// SAGPool_22514218566433
// MI455X (gfx1250) — matrix operand order repaired and checked by reading, not compiled, not run on hardware
//
#include <hip/hip_runtime.h>
#include <math.h>

// ---------------- problem constants (from reference) ----------------
#define NB      256                 // graphs per batch
#define N0      512                 // nodes per graph initially
#define EPG     8192                // edges per graph (E/B)
#define E_TOT   (NB * EPG)          // 2,097,152 edges
#define F_INF   160
#define HD      64
#define K1      57                  // ceil(0.11*512)
#define K2      7                   // ceil(0.11*57)

typedef __attribute__((ext_vector_type(16))) _Float16 v16h;
typedef __attribute__((ext_vector_type(8)))  float    v8f;

// =====================================================================
// 1) Per-graph mean aggregation: agg[d] = mean_{e: dst=d} x[src[e]]
//    One workgroup per (graph, feature-chunk). Scatter into LDS with
//    ds_add_f32 atomics, normalize by degree in LDS, single HBM write.
//    FC templated so the hot p/FC becomes a mul-shift.
// =====================================================================
template<int FC>
__global__ void __launch_bounds__(512)
k_agg(const float* __restrict__ x, const int* __restrict__ src,
      const int* __restrict__ dst, const int* __restrict__ valid,
      float* __restrict__ aggn, int n_per, int F)
{
    extern __shared__ float smem[];        // n_per*FC (agg) + n_per (deg)
    float* aggl = smem;
    float* degl = smem + n_per * FC;

    const int g   = blockIdx.x;
    const int f0  = blockIdx.y * FC;
    const int tid = threadIdx.x, nt = blockDim.x;

    for (int i = tid; i < n_per * FC + n_per; i += nt) smem[i] = 0.f;
    __syncthreads();

    const int e0       = g * EPG;
    const int nodebase = g * n_per;
    const int total    = EPG * FC;
    // (edge, feature) pairs: consecutive lanes take consecutive features
    // of the same edge -> coalesced gather of x[src].
    for (int p = tid; p < total; p += nt) {
        int e  = p / FC;                       // constant divisor -> mulhi
        int f  = p - e * FC;
        int ei = e0 + e;
        if (!valid || valid[ei]) {
            int s = src[ei];
            int d = dst[ei] - nodebase;        // local node id
            atomicAdd(&aggl[d * FC + f], x[(long)s * F + f0 + f]); // ds_add_f32
            if (f == 0) atomicAdd(&degl[d], 1.0f);
        }
    }
    __syncthreads();

    for (int i = tid; i < n_per * FC; i += nt) {
        int node = i / FC;
        int f    = i - node * FC;
        aggn[(long)(nodebase + node) * F + f0 + f] =
            aggl[i] / fmaxf(degl[node], 1.0f);
    }
}

// =====================================================================
// 2) Fused GraphConv GEMM: Y[M,64] = act(A0@W0 + A1@W1 + bias)
//    f16 WMMA 16x16x32, f32 accumulate. Block = 128 thr (4 waves);
//    block owns a 16-row M tile (M is a multiple of 16 by construction),
//    wave w owns output cols [16w,16w+16). W staged once per pass into
//    LDS (f16, transposed); A tile staged per-kt via b128 loads.
// =====================================================================
template<int K>
__global__ void __launch_bounds__(128)
k_gemm_bias_relu(const float* __restrict__ A0, const float* __restrict__ W0,
                 const float* __restrict__ A1, const float* __restrict__ W1,
                 const float* __restrict__ bias, float* __restrict__ Y,
                 int do_relu)
{
    __shared__ _Float16 Ws[64 * K];            // transposed: Ws[n*K + k]
    __shared__ _Float16 As[16][40];            // 16x32 f16 A tile (padded)
    const int m0   = blockIdx.x * 16;
    const int wave = threadIdx.x >> 5;
    const int lane = threadIdx.x & 31;
    const int n0   = wave * 16;
    const int col  = lane & 15;                // N index (and A row)
    const int hi   = lane >> 4;                // half-wave selector
    constexpr int NK = K / 32;

    const int sr  = threadIdx.x >> 3;          // staging row 0..15
    const int sc4 = (threadIdx.x & 7) << 2;    // staging col group

    v8f acc = {};
    for (int pass = 0; pass < 2; ++pass) {
        const float* A = pass ? A1 : A0;
        const float* W = pass ? W1 : W0;
        if (!A) break;                         // uniform (only lin1 passes null)
        __syncthreads();                       // Ws reuse across passes
        // stage W -> LDS f16, transposed so B fragments read contiguous K
        for (int t = threadIdx.x; t < K * 64; t += 128) {
            int k = t >> 6, n = t & 63;
            Ws[n * K + k] = (_Float16)W[t];
        }
#pragma unroll
        for (int kt = 0; kt < NK; ++kt) {
            __syncthreads();
            {   // stage 16x32 A tile as f16 (one b128 load per thread)
                const float4 v = *(const float4*)(A + (long)(m0 + sr) * K +
                                                  kt * 32 + sc4);
                As[sr][sc4 + 0] = (_Float16)v.x;
                As[sr][sc4 + 1] = (_Float16)v.y;
                As[sr][sc4 + 2] = (_Float16)v.z;
                As[sr][sc4 + 3] = (_Float16)v.w;
            }
            if (kt + 1 < NK)   // pull next tile toward L2/L0 (global_prefetch_b8)
                __builtin_prefetch(A + (long)(m0 + sr) * K + (kt + 1) * 32 + sc4,
                                   0, 3);
            __syncthreads();
            // A fragment: 16-bit A 16x32 layout (ISA 7.12.2)
            //  lanes 0-15: j<8 -> K=j,  j>=8 -> K=j+8 ; lanes 16-31: +8
            v16h a, bW;
#pragma unroll
            for (int j = 0; j < 16; ++j) {
                int ka = j + ((j >= 8) ? 8 : 0) + (hi ? 8 : 0);
                a[j] = As[col][ka];
                // B fragment: 32x16, same K order as A (j<8 -> K=j, j>=8 -> K=j+8; lanes 16-31: +8)
                int kb = kt * 32 + j + ((j >= 8) ? 8 : 0) + (hi ? 8 : 0);
                bW[j] = Ws[(n0 + col) * K + kb];
            }
            acc = __builtin_amdgcn_wmma_f32_16x16x32_f16(
                      false, a, false, bW, (short)0, acc, false, false);
        }
    }

    float bv = bias ? bias[n0 + col] : 0.f;
#pragma unroll
    for (int r = 0; r < 8; ++r) {              // C/D: VGPR r -> M = r + 8*hi
        float v = acc[r] + bv;
        if (do_relu) v = fmaxf(v, 0.f);
        Y[(long)(m0 + r + hi * 8) * HD + n0 + col] = v;
    }
}

// =====================================================================
// 3) global_mean_pool into the JumpingKnowledge concat buffer [NB, 6*64]
// =====================================================================
__global__ void k_gmp(const float* __restrict__ y, float* __restrict__ xs_cat,
                      int n_per, int slot)
{
    int g = blockIdx.x, h = threadIdx.x;       // 64 threads
    float s = 0.f;
    long base = (long)g * n_per * HD;
    for (int i = 0; i < n_per; ++i) s += y[base + (long)i * HD + h];
    xs_cat[g * (6 * HD) + slot * HD + h] = s / (float)n_per;
}

// =====================================================================
// 4) SAG score: score[n] = agg[n].Wr + x[n].Wl + b   (H -> 1 GraphConv)
// =====================================================================
__global__ void k_score(const float* __restrict__ aggn, const float* __restrict__ xin,
                        const float* __restrict__ Wr, const float* __restrict__ Wl,
                        const float* __restrict__ bb, float* __restrict__ score, int n)
{
    int i = blockIdx.x * blockDim.x + threadIdx.x;
    if (i >= n) return;
    float s = bb[0];
#pragma unroll 8
    for (int h = 0; h < HD; ++h)
        s += aggn[(long)i * HD + h] * Wr[h] + xin[(long)i * HD + h] * Wl[h];
    score[i] = s;
}

// =====================================================================
// 5) Top-k select per graph (stable rank, matches lax.top_k tie order),
//    gate by tanh(score), emit newpos for edge remap.
// =====================================================================
__global__ void __launch_bounds__(512)
k_sagpool(const float* __restrict__ score, const float* __restrict__ xin,
          float* __restrict__ xout, int* __restrict__ newpos, int n_per, int k)
{
    __shared__ float s[512];
    int g = blockIdx.x, i = threadIdx.x;
    if (i < n_per) s[i] = score[g * n_per + i];
    __syncthreads();
    if (i >= n_per) return;
    float si = s[i];
    int r = 0;
    for (int j = 0; j < n_per; ++j) {
        float sj = s[j];
        r += (sj > si) || (sj == si && j < i);
    }
    int node = g * n_per + i;
    if (r < k) {
        int nn = g * k + r;
        newpos[node] = nn;
        float t = tanhf(si);
        for (int h = 0; h < HD; ++h)
            xout[(long)nn * HD + h] = xin[(long)node * HD + h] * t;
    } else {
        newpos[node] = -1;
    }
}

// =====================================================================
// 6) Edge remap (safe in place: element-wise)
// =====================================================================
__global__ void k_remap(const int* __restrict__ src_in, const int* __restrict__ dst_in,
                        const int* __restrict__ valid_in, const int* __restrict__ newpos,
                        int* __restrict__ src_out, int* __restrict__ dst_out,
                        int* __restrict__ valid_out)
{
    int e = blockIdx.x * blockDim.x + threadIdx.x;
    if (e >= E_TOT) return;
    int v  = valid_in ? valid_in[e] : 1;
    int ns = newpos[src_in[e]];
    int nd = newpos[dst_in[e]];
    int nv = v && (ns >= 0) && (nd >= 0);
    src_out[e]   = nv ? ns : 0;
    dst_out[e]   = nv ? nd : 0;
    valid_out[e] = nv;
}

// =====================================================================
// 7) Head: logits = h1 @ W[64,2] + b ; out = log_softmax
// =====================================================================
__global__ void k_head(const float* __restrict__ h1, const float* __restrict__ W,
                       const float* __restrict__ b, float* __restrict__ out)
{
    int row = blockIdx.x * blockDim.x + threadIdx.x;
    if (row >= NB) return;
    float z0 = b[0], z1 = b[1];
#pragma unroll 8
    for (int i = 0; i < HD; ++i) {
        float v = h1[row * HD + i];
        z0 += v * W[i * 2 + 0];
        z1 += v * W[i * 2 + 1];
    }
    float m   = fmaxf(z0, z1);
    float lse = m + logf(expf(z0 - m) + expf(z1 - m));
    out[row * 2 + 0] = z0 - lse;
    out[row * 2 + 1] = z1 - lse;
}

// =====================================================================
// launch helpers (template dispatch)
// =====================================================================
static void launch_agg(const float* x, const int* s, const int* d, const int* v,
                       float* aggn, int n_per, int F, int FC, hipStream_t st)
{
    size_t sm = (size_t)(n_per * FC + n_per) * 4;
    dim3 grid(NB, F / FC);
    if (FC == 20)      k_agg<20><<<grid, 512, sm, st>>>(x, s, d, v, aggn, n_per, F);
    else if (FC == 16) k_agg<16><<<grid, 512, sm, st>>>(x, s, d, v, aggn, n_per, F);
    else               k_agg<64><<<grid, 512, sm, st>>>(x, s, d, v, aggn, n_per, F);
}

static void launch_gemm(const float* A0, const float* W0, const float* A1,
                        const float* W1, const float* bias, float* Y,
                        int M, int K, hipStream_t st)
{
    if (K == 160)
        k_gemm_bias_relu<160><<<M / 16, 128, 0, st>>>(A0, W0, A1, W1, bias, Y, 1);
    else if (K == 64)
        k_gemm_bias_relu<64><<<M / 16, 128, 0, st>>>(A0, W0, A1, W1, bias, Y, 1);
    else
        k_gemm_bias_relu<384><<<M / 16, 128, 0, st>>>(A0, W0, A1, W1, bias, Y, 1);
}

// =====================================================================
// host-side orchestration
// =====================================================================
extern "C" void kernel_launch(void* const* d_in, const int* in_sizes, int n_in,
                              void* d_out, int out_size, void* d_ws, size_t ws_size,
                              hipStream_t stream)
{
    const float* x        = (const float*)d_in[0];
    const int*   ei       = (const int*)  d_in[1];
    const float* c1_Wr    = (const float*)d_in[2];
    const float* c1_Wl    = (const float*)d_in[3];
    const float* c1_b     = (const float*)d_in[4];
    const float* cs_Wr    = (const float*)d_in[5];    // [5,64,64]
    const float* cs_Wl    = (const float*)d_in[6];
    const float* cs_b     = (const float*)d_in[7];    // [5,64]
    const float* p_Wr     = (const float*)d_in[8];    // [2,64,1]
    const float* p_Wl     = (const float*)d_in[9];    // [2,64]
    const float* p_b      = (const float*)d_in[10];   // [2,1]
    const float* l1_W     = (const float*)d_in[11];   // [384,64]
    const float* l1_b     = (const float*)d_in[12];
    const float* l2_W     = (const float*)d_in[13];   // [64,2]
    const float* l2_b     = (const float*)d_in[14];
    float*       out      = (float*)d_out;

    const int* src0 = ei;
    const int* dst0 = ei + E_TOT;

    // ---- carve workspace ----
    char* w = (char*)d_ws;
    auto carve = [&](size_t bytes) -> void* {
        void* p = (void*)w;
        w += (bytes + 255) & ~(size_t)255;
        return p;
    };
    const int NTOT = NB * N0;                         // 131072
    float* aggn   = (float*)carve((size_t)NTOT * F_INF * 4);
    float* nb0    = (float*)carve((size_t)NTOT * HD * 4);
    float* nb1    = (float*)carve((size_t)NTOT * HD * 4);
    float* nb2    = (float*)carve((size_t)NTOT * HD * 4);
    float* score  = (float*)carve((size_t)NTOT * 4);
    int*   newpos = (int*)  carve((size_t)NTOT * 4);
    int*   esrc   = (int*)  carve((size_t)E_TOT * 4);
    int*   edst   = (int*)  carve((size_t)E_TOT * 4);
    int*   evalid = (int*)  carve((size_t)E_TOT * 4);
    float* xs_cat = (float*)carve((size_t)NB * 6 * HD * 4);
    float* h1     = (float*)carve((size_t)NB * HD * 4);
    (void)ws_size; (void)n_in; (void)in_sizes; (void)out_size;

    // ---- conv1: F_IN -> H ----
    launch_agg(x, src0, dst0, nullptr, aggn, N0, F_INF, 20, stream);
    launch_gemm(aggn, c1_Wr, x, c1_Wl, c1_b, nb0, NTOT, F_INF, stream);   // y1=nb0
    k_gmp<<<NB, HD, 0, stream>>>(nb0, xs_cat, N0, 0);

    // ---- conv i=0 ----
    launch_agg(nb0, src0, dst0, nullptr, aggn, N0, HD, 16, stream);
    launch_gemm(aggn, cs_Wr + 0 * 4096, nb0, cs_Wl + 0 * 4096, cs_b + 0 * HD,
                nb1, NTOT, HD, stream);                                   // y2=nb1
    k_gmp<<<NB, HD, 0, stream>>>(nb1, xs_cat, N0, 1);

    // ---- SAG pool 0: 512 -> 57 ----
    launch_agg(nb1, src0, dst0, nullptr, aggn, N0, HD, 16, stream);
    k_score<<<(NTOT + 255) / 256, 256, 0, stream>>>(aggn, nb1, p_Wr, p_Wl, p_b,
                                                    score, NTOT);
    k_sagpool<<<NB, 512, 0, stream>>>(score, nb1, nb2, newpos, N0, K1);   // x3=nb2
    k_remap<<<(E_TOT + 255) / 256, 256, 0, stream>>>(src0, dst0, nullptr, newpos,
                                                     esrc, edst, evalid);
    const int N1 = NB * K1;                                               // 14592

    // ---- conv i=1 ----
    launch_agg(nb2, esrc, edst, evalid, aggn, K1, HD, 64, stream);
    launch_gemm(aggn, cs_Wr + 1 * 4096, nb2, cs_Wl + 1 * 4096, cs_b + 1 * HD,
                nb0, N1, HD, stream);                                     // y3=nb0
    k_gmp<<<NB, HD, 0, stream>>>(nb0, xs_cat, K1, 2);

    // ---- conv i=2 ----
    launch_agg(nb0, esrc, edst, evalid, aggn, K1, HD, 64, stream);
    launch_gemm(aggn, cs_Wr + 2 * 4096, nb0, cs_Wl + 2 * 4096, cs_b + 2 * HD,
                nb1, N1, HD, stream);                                     // y4=nb1
    k_gmp<<<NB, HD, 0, stream>>>(nb1, xs_cat, K1, 3);

    // ---- SAG pool 1: 57 -> 7 ----
    launch_agg(nb1, esrc, edst, evalid, aggn, K1, HD, 64, stream);
    k_score<<<(N1 + 255) / 256, 256, 0, stream>>>(aggn, nb1, p_Wr + HD, p_Wl + HD,
                                                  p_b + 1, score, N1);
    k_sagpool<<<NB, 512, 0, stream>>>(score, nb1, nb2, newpos, K1, K2);   // x5=nb2
    k_remap<<<(E_TOT + 255) / 256, 256, 0, stream>>>(esrc, edst, evalid, newpos,
                                                     esrc, edst, evalid); // in place
    const int N2 = NB * K2;                                               // 1792

    // ---- conv i=3 ----
    launch_agg(nb2, esrc, edst, evalid, aggn, K2, HD, 64, stream);
    launch_gemm(aggn, cs_Wr + 3 * 4096, nb2, cs_Wl + 3 * 4096, cs_b + 3 * HD,
                nb0, N2, HD, stream);                                     // y5=nb0
    k_gmp<<<NB, HD, 0, stream>>>(nb0, xs_cat, K2, 4);

    // ---- conv i=4 ----
    launch_agg(nb0, esrc, edst, evalid, aggn, K2, HD, 64, stream);
    launch_gemm(aggn, cs_Wr + 4 * 4096, nb0, cs_Wl + 4 * 4096, cs_b + 4 * HD,
                nb1, N2, HD, stream);                                     // y6=nb1
    k_gmp<<<NB, HD, 0, stream>>>(nb1, xs_cat, K2, 5);

    // ---- head: lin1 (WMMA) + lin2 + log_softmax ----
    launch_gemm(xs_cat, l1_W, nullptr, nullptr, l1_b, h1, NB, 6 * HD, stream);
    k_head<<<1, NB, 0, stream>>>(h1, l2_W, l2_b, out);
}
